// GCN_28166395527417
// MI455X (gfx1250) — matrix operand order repaired and checked by reading, not compiled, not run on hardware
//
#include <hip/hip_runtime.h>
#include <cstddef>

#define BN_EPS 1e-5f

typedef __attribute__((ext_vector_type(16))) _Float16 v16h;
typedef __attribute__((ext_vector_type(8)))  float    v8f;

// ---------------------------------------------------------------------------
// Native no-return fp32 atomic add (global_atomic_add_f32)
// ---------------------------------------------------------------------------
__device__ __forceinline__ void fadd32(float* p, float v) {
    unsafeAtomicAdd(p, v);
}

// ---------------------------------------------------------------------------
// Degree / normalization prep
// ---------------------------------------------------------------------------
__global__ void k_deg_init(float* __restrict__ deg, int N) {
    int i = blockIdx.x * blockDim.x + threadIdx.x;
    if (i < N) deg[i] = 1.0f;                      // +1 for self loop
}

__global__ void k_deg_count(const int* __restrict__ dst, float* __restrict__ deg, int E) {
    int i = blockIdx.x * blockDim.x + threadIdx.x;
    if (i < E) fadd32(&deg[dst[i]], 1.0f);
}

__global__ void k_norm(const float* __restrict__ deg, float* __restrict__ dinv,
                       float* __restrict__ invdeg, int N) {
    int i = blockIdx.x * blockDim.x + threadIdx.x;
    if (i < N) {
        float d = deg[i];
        dinv[i]   = rsqrtf(d);
        invdeg[i] = 1.0f / d;
    }
}

// ---------------------------------------------------------------------------
// Pre-pack W[K x F] (f32, row-major) into per-lane WMMA B-fragment order, f16.
// Fragment layout used here (contiguous k; wave32, 16-bit B 32x16): lane L holds column
// L%16; vector element pair (2j, 2j+1) covers K = kt*32 + 16*(L/16) + 2j (+1).
// One thread per (tileN, kTile, lane) emits its 16 contiguous f16.
// ---------------------------------------------------------------------------
__global__ void k_packW(const float* __restrict__ W, _Float16* __restrict__ Wp,
                        int K, int F, int NT) {
    int tid = blockIdx.x * blockDim.x + threadIdx.x;
    int Kt = K >> 5;
    int total = NT * Kt * 32;
    if (tid >= total) return;
    int lane = tid & 31;
    int kt   = (tid >> 5) % Kt;
    int tn   = (tid >> 5) / Kt;
    int ml   = lane & 15;
    int half = lane >> 4;
    int col  = tn * 16 + ml;
    int colc = col < F ? col : F - 1;              // clamp (F=40 padded to 48)
    _Float16* out = Wp + (size_t)tid * 16;
    #pragma unroll
    for (int j = 0; j < 8; ++j) {
        int kb = kt * 32 + half * 16 + 2 * j;
        out[2 * j]     = (_Float16)W[(size_t)kb       * F + colc];
        out[2 * j + 1] = (_Float16)W[(size_t)(kb + 1) * F + colc];
    }
}

// ---------------------------------------------------------------------------
// WMMA GEMM:  H[N x F] = act(A[N x K]) * W[K x F]
// act = identity, or fused BatchNorm(eval)+ReLU on A elements (per-K channel),
// so the post-BN activation never hits memory. One wave32 computes one 16x16
// output tile via K/32 v_wmma_f32_16x16x32_f16. B fragments come pre-packed.
// ---------------------------------------------------------------------------
template<int K, bool FUSE_BN>
__global__ __launch_bounds__(128)
void k_gemm(const float* __restrict__ A, const _Float16* __restrict__ Wp,
            const float* __restrict__ bm, const float* __restrict__ bv,
            const float* __restrict__ bg, const float* __restrict__ bb,
            float* __restrict__ H, int Nrows, int F, int NT) {
    const int wavesPerBlock = blockDim.x >> 5;
    const int wave = blockIdx.x * wavesPerBlock + (threadIdx.x >> 5);
    const int MT = (Nrows + 15) >> 4;
    if (wave >= MT * NT) return;                   // wave-uniform exit (EXEC stays all-1s)

    const int lane = threadIdx.x & 31;
    const int tm = wave / NT, tn = wave % NT;
    const int ml   = lane & 15;
    const int half = lane >> 4;

    const int row  = tm * 16 + ml;
    const int rowc = row < Nrows ? row : Nrows - 1;          // clamped A row
    const int col  = tn * 16 + ml;

    const float* __restrict__ Arow = A + (size_t)rowc * K;
    const v16h* __restrict__ Bfrag =
        reinterpret_cast<const v16h*>(Wp) + ((size_t)tn * (K >> 5) * 32 + lane);

    v8f acc = {};
    #pragma unroll
    for (int kt = 0; kt < K / 32; ++kt) {
        v16h af;
        #pragma unroll
        for (int j = 0; j < 8; ++j) {
            // A fragment: lane holds row ml; VGPR j -> K pair 16*half + 2j (matches packed B)
            int ka = kt * 32 + half * 16 + 2 * j;
            float a0 = Arow[ka];
            float a1 = Arow[ka + 1];
            if (FUSE_BN) {
                a0 = fmaxf((a0 - bm[ka])     * rsqrtf(bv[ka]     + BN_EPS) * bg[ka]     + bb[ka],     0.0f);
                a1 = fmaxf((a1 - bm[ka + 1]) * rsqrtf(bv[ka + 1] + BN_EPS) * bg[ka + 1] + bb[ka + 1], 0.0f);
            }
            af[2 * j]     = (_Float16)a0;
            af[2 * j + 1] = (_Float16)a1;
        }
        const v16h bf = Bfrag[(size_t)kt * 32];
        acc = __builtin_amdgcn_wmma_f32_16x16x32_f16(
            /*neg_a=*/false, af, /*neg_b=*/false, bf,
            /*c_mod=*/(short)0, acc, /*reuse_a=*/false, /*reuse_b=*/false);
    }

    #pragma unroll
    for (int r = 0; r < 8; ++r) {
        int orow = tm * 16 + r + half * 8;                   // C/D layout
        if (orow < Nrows && col < F) H[(size_t)orow * F + col] = acc[r];
    }
}

// ---------------------------------------------------------------------------
// agg = h * inv_deg[row] + bias[col]   (fuses self-loop term + bias + zero-init)
// ---------------------------------------------------------------------------
__global__ void k_selfbias(const float* __restrict__ h, const float* __restrict__ invdeg,
                           const float* __restrict__ bias, float* __restrict__ agg,
                           int N, int F) {
    size_t tid = (size_t)blockIdx.x * blockDim.x + threadIdx.x;
    size_t total = (size_t)N * F;
    if (tid >= total) return;
    int row = (int)(tid / F);
    int colf = (int)(tid - (size_t)row * F);
    agg[tid] = h[tid] * invdeg[row] + bias[colf];
}

// ---------------------------------------------------------------------------
// Edge scatter: agg[dst] += dinv[src]*dinv[dst] * h[src], float4 per thread
// ---------------------------------------------------------------------------
__global__ void k_scatter(const float* __restrict__ h, float* __restrict__ agg,
                          const int* __restrict__ src, const int* __restrict__ dst,
                          const float* __restrict__ dinv, int E, int F, int groups) {
    size_t tid = (size_t)blockIdx.x * blockDim.x + threadIdx.x;
    if (tid >= (size_t)E * groups) return;
    int e = (int)(tid / groups);
    int f = (int)(tid - (size_t)e * groups) * 4;
    int s = src[e], d = dst[e];
    float w = dinv[s] * dinv[d];
    const float4 hv = *reinterpret_cast<const float4*>(h + (size_t)s * F + f);
    float* p = agg + (size_t)d * F + f;
    fadd32(p + 0, hv.x * w);
    fadd32(p + 1, hv.y * w);
    fadd32(p + 2, hv.z * w);
    fadd32(p + 3, hv.w * w);
}

// ---------------------------------------------------------------------------
// Row-wise log_softmax over 40 classes (in-place safe: row cached in regs)
// ---------------------------------------------------------------------------
__global__ void k_logsoftmax40(const float* __restrict__ in, float* __restrict__ out, int N) {
    int row = blockIdx.x * blockDim.x + threadIdx.x;
    if (row >= N) return;
    float v[40];
    float mx = -3.402823e38f;
    const float* r = in + (size_t)row * 40;
    #pragma unroll
    for (int c = 0; c < 40; ++c) { v[c] = r[c]; mx = fmaxf(mx, v[c]); }
    float s = 0.0f;
    #pragma unroll
    for (int c = 0; c < 40; ++c) s += __expf(v[c] - mx);
    float lse = mx + __logf(s);
    float* o = out + (size_t)row * 40;
    #pragma unroll
    for (int c = 0; c < 40; ++c) o[c] = v[c] - lse;
}

// ---------------------------------------------------------------------------
extern "C" void kernel_launch(void* const* d_in, const int* in_sizes, int n_in,
                              void* d_out, int out_size, void* d_ws, size_t ws_size,
                              hipStream_t stream) {
    const float* x   = (const float*)d_in[0];
    const int*   src = (const int*)  d_in[1];
    const int*   dst = (const int*)  d_in[2];
    const float* W1  = (const float*)d_in[3];
    const float* b1  = (const float*)d_in[4];
    const float* g1  = (const float*)d_in[5];
    const float* be1 = (const float*)d_in[6];
    const float* m1  = (const float*)d_in[7];
    const float* v1  = (const float*)d_in[8];
    const float* W2  = (const float*)d_in[9];
    const float* b2  = (const float*)d_in[10];
    const float* g2  = (const float*)d_in[11];
    const float* be2 = (const float*)d_in[12];
    const float* m2  = (const float*)d_in[13];
    const float* v2  = (const float*)d_in[14];
    const float* W3  = (const float*)d_in[15];
    const float* b3  = (const float*)d_in[16];
    float* out = (float*)d_out;

    const int N = in_sizes[0] / 128;
    const int E = in_sizes[1];

    // Workspace layout
    float* ws     = (float*)d_ws;
    float* deg    = ws;
    float* dinv   = deg  + N;
    float* invdeg = dinv + N;
    float* H      = invdeg + N + 2;            // pad to keep 16-float alignment
    float* AG     = H + (size_t)N * 256;       // N x 256 (aggregated)
    float* H3     = AG + (size_t)N * 256;      // N x 40
    // packed f16 B fragments (32-byte aligned: offsets are multiples of 16 floats)
    _Float16* Wp1 = (_Float16*)(H3 + (size_t)N * 40);      // 16 tiles * 4 kt * 32 * 16
    _Float16* Wp2 = Wp1 + (size_t)16 * 4 * 32 * 16;        // 16 tiles * 8 kt * 32 * 16
    _Float16* Wp3 = Wp2 + (size_t)16 * 8 * 32 * 16;        //  3 tiles * 8 kt * 32 * 16

    auto blocks = [](size_t n, int bs) { return dim3((unsigned)((n + bs - 1) / bs)); };

    // --- normalization prep + weight packing ---
    k_deg_init <<<blocks(N, 256), 256, 0, stream>>>(deg, N);
    k_deg_count<<<blocks(E, 256), 256, 0, stream>>>(dst, deg, E);
    k_norm     <<<blocks(N, 256), 256, 0, stream>>>(deg, dinv, invdeg, N);
    k_packW    <<<blocks(16 * 4 * 32, 256), 256, 0, stream>>>(W1, Wp1, 128, 256, 16);
    k_packW    <<<blocks(16 * 8 * 32, 256), 256, 0, stream>>>(W2, Wp2, 256, 256, 16);
    k_packW    <<<blocks( 3 * 8 * 32, 256), 256, 0, stream>>>(W3, Wp3, 256,  40,  3);

    const int MT = (N + 15) / 16;
    auto gemmBlocks = [&](int NT) { return dim3((unsigned)(((size_t)MT * NT + 3) / 4)); };

    // --- layer 1: H = x @ W1 ---
    k_gemm<128, false><<<gemmBlocks(16), 128, 0, stream>>>(
        x, Wp1, nullptr, nullptr, nullptr, nullptr, H, N, 256, 16);
    k_selfbias<<<blocks((size_t)N * 256, 256), 256, 0, stream>>>(H, invdeg, b1, AG, N, 256);
    k_scatter <<<blocks((size_t)E * 64, 256), 256, 0, stream>>>(H, AG, src, dst, dinv, E, 256, 64);

    // --- layer 2: H = relu(bn1(AG)) @ W2   (BN+ReLU fused into A-load) ---
    k_gemm<256, true><<<gemmBlocks(16), 128, 0, stream>>>(
        AG, Wp2, m1, v1, g1, be1, H, N, 256, 16);
    k_selfbias<<<blocks((size_t)N * 256, 256), 256, 0, stream>>>(H, invdeg, b2, AG, N, 256);
    k_scatter <<<blocks((size_t)E * 64, 256), 256, 0, stream>>>(H, AG, src, dst, dinv, E, 256, 64);

    // --- layer 3: H3 = relu(bn2(AG)) @ W3 ---
    k_gemm<256, true><<<gemmBlocks(3), 128, 0, stream>>>(
        AG, Wp3, m2, v2, g2, be2, H3, N, 40, 3);
    k_selfbias<<<blocks((size_t)N * 40, 256), 256, 0, stream>>>(H3, invdeg, b3, out, N, 40);
    k_scatter <<<blocks((size_t)E * 10, 256), 256, 0, stream>>>(H3, out, src, dst, dinv, E, 40, 10);

    // --- log_softmax ---
    k_logsoftmax40<<<blocks(N, 256), 256, 0, stream>>>(out, out, N);
}
